// TransformerAttention_6622839570513
// MI455X (gfx1250) — compile-verified, matrix operand order checked by reading, not run on hardware
//
#include <hip/hip_runtime.h>

#define HID   1024
#define SEQ   2048
#define NB    4
#define HEADS 16
#define HD    64

typedef __attribute__((ext_vector_type(8)))  __bf16 v8bf;
typedef __attribute__((ext_vector_type(16))) __bf16 v16bf;
typedef __attribute__((ext_vector_type(8)))  float  v8f;
typedef __attribute__((ext_vector_type(4)))  unsigned int u32x4;
typedef __attribute__((ext_vector_type(8)))  int i32x8;
typedef __attribute__((ext_vector_type(4)))  int i32x4;

#if defined(__gfx1250__) && __has_builtin(__builtin_amdgcn_tensor_load_to_lds)
#define USE_TDM 1
#else
#define USE_TDM 0
#endif

__device__ __forceinline__ unsigned short f2bf(float f) {
  union { float f; unsigned u; } v; v.f = f;
  unsigned r = v.u + 0x7FFFu + ((v.u >> 16) & 1u);   // round-to-nearest-even
  return (unsigned short)(r >> 16);
}

__device__ __forceinline__ v8f wmma_bf16(v16bf a, v16bf b, v8f c) {
  return __builtin_amdgcn_wmma_f32_16x16x32_bf16(false, a, false, b, (short)0, c,
                                                 false, false);
}

#if USE_TDM
// Issue a TDM 2D tile load (bf16 elements) into LDS. Descriptor per
// CDNA5 ISA: group0 = {flags/count, lds_addr,
// global_addr, type=2}; group1 = {data_size, tensor dims, tile dims, stride}.
// Tracked by TENSORcnt. 6-arg builtin form (clang-23 / therock-10.0 lane).
__device__ __forceinline__ void tdm_load_2d_bf16(
    unsigned lds_off, const unsigned short* gptr,
    unsigned tensor_d0, unsigned tensor_d1, unsigned stride0,
    unsigned tile_d0, unsigned tile_d1) {
  unsigned long long ga = (unsigned long long)(size_t)gptr;
  u32x4 g0;
  g0[0] = 1u;                                          // count=1, user mode
  g0[1] = lds_off;                                     // lds_addr (bytes)
  g0[2] = (unsigned)(ga & 0xFFFFFFFFu);                // global_addr[31:0]
  g0[3] = (unsigned)((ga >> 32) & 0x1FFFFFFu) | (2u << 30);  // addr hi | type=2
  i32x8 g1;
  g1[0] = (int)(1u << 16);                             // data_size=1 -> 2 bytes
  g1[1] = (int)((tensor_d0 & 0xFFFFu) << 16);          // dim0[15:0]
  g1[2] = (int)((tensor_d0 >> 16) | ((tensor_d1 & 0xFFFFu) << 16));
  g1[3] = (int)((tensor_d1 >> 16) | (tile_d0 << 16));  // dim1 hi | tile_dim0
  g1[4] = (int)(tile_d1 & 0xFFFFu);                    // tile_dim1, tile_dim2=0
  g1[5] = (int)stride0;                                // dim0_stride[31:0]
  g1[6] = 0;                                           // stride hi, dim1_stride
  g1[7] = 0;
  i32x4 gz4 = {0, 0, 0, 0};
  i32x8 gz8 = {0, 0, 0, 0, 0, 0, 0, 0};
  __builtin_amdgcn_tensor_load_to_lds(g0, g1, gz4, gz4, gz8, 0);
}
__device__ __forceinline__ unsigned lds_offset_of(const void* p) {
  return (unsigned)(unsigned long long)(size_t)p;       // low 32b = LDS offset
}
#endif

// Load a 16x32 bf16 A/B fragment from a row-major tile (ld = LD bf16 elems).
// Lanes 0-15 rows M=0..15 (K 0..7,16..23), lanes 16-31 same rows (K 8..15,
// 24..31). Two b128 loads per lane.
template <int LD>
__device__ __forceinline__ v16bf load_frag(const unsigned short* base, int row) {
  const int lane = threadIdx.x & 31;
  const int m    = lane & 15;
  const int hh   = (lane >> 4) & 1;
  const unsigned short* p = base + (size_t)(row + m) * LD + hh * 8;
  v8bf lo = *(const v8bf*)p;
  v8bf hi = *(const v8bf*)(p + 16);
  v16bf r;
#pragma unroll
  for (int i = 0; i < 8; ++i) { r[i] = lo[i]; r[i + 8] = hi[i]; }
  return r;
}

// ---------------------------------------------------------------------------
// Kernel 1: fused QKV projection.  X[8192,1024] (fp32) x W[1024,1024] (fp32)
// -> bf16 Q (scaled by 1/8) / K row-major per head, V stored transposed.
// ---------------------------------------------------------------------------
__global__ __launch_bounds__(256) void qkv_kernel(
    const float* __restrict__ X,
    const float* __restrict__ Wq, const float* __restrict__ bq,
    const float* __restrict__ Wk, const float* __restrict__ bk,
    const float* __restrict__ Wv, const float* __restrict__ bv,
    unsigned short* __restrict__ qbuf, unsigned short* __restrict__ kbuf,
    unsigned short* __restrict__ vtbuf) {
  __shared__ unsigned short sA[128 * 32];
  __shared__ unsigned short sB[128 * 32];
  const int z  = blockIdx.z;
  const float* W    = (z == 0) ? Wq : (z == 1) ? Wk : Wv;
  const float* bias = (z == 0) ? bq : (z == 1) ? bk : bv;
  const int m0 = blockIdx.x * 128;
  const int n0 = blockIdx.y * 128;
  const int tid  = threadIdx.x;
  const int wave = tid >> 5;
  const int wm   = wave & 3;
  const int wn   = wave >> 2;

  v8f zero = {};
  v8f acc[2][4];
#pragma unroll
  for (int i = 0; i < 2; ++i)
#pragma unroll
    for (int j = 0; j < 4; ++j) acc[i][j] = zero;

  for (int k0 = 0; k0 < HID; k0 += 32) {
    __syncthreads();
    {   // A tile: 128x32 fp32 -> bf16
      int r = tid >> 1, c = (tid & 1) * 16;
      const float* src = X + (size_t)(m0 + r) * HID + k0 + c;
      if (k0 + 32 < HID) __builtin_prefetch(src + 32, 0, 1);
      unsigned short tmp[16];
#pragma unroll
      for (int i = 0; i < 16; ++i) tmp[i] = f2bf(src[i]);
      *(uint4*)(sA + r * 32 + c)     = *(uint4*)tmp;
      *(uint4*)(sA + r * 32 + c + 8) = *(uint4*)(tmp + 8);
    }
    {   // B tile: W[k0:k0+32, n0:n0+128] transposed into [n][k]
      int k = tid >> 3, nb = (tid & 7) * 16;
      const float* src = W + (size_t)(k0 + k) * HID + n0 + nb;
      if (k0 + 32 < HID) __builtin_prefetch(src + 32 * HID, 0, 1);
#pragma unroll
      for (int i = 0; i < 16; ++i) sB[(nb + i) * 32 + k] = f2bf(src[i]);
    }
    __syncthreads();
    v16bf a0 = load_frag<32>(sA, wm * 32);
    v16bf a1 = load_frag<32>(sA, wm * 32 + 16);
#pragma unroll
    for (int j = 0; j < 4; ++j) {
      v16bf bf = load_frag<32>(sB, wn * 64 + j * 16);
      acc[0][j] = wmma_bf16(a0, bf, acc[0][j]);
      acc[1][j] = wmma_bf16(a1, bf, acc[1][j]);
    }
  }

  const int lane = tid & 31;
  const int nloc = lane & 15;
  const int mofs = ((lane >> 4) & 1) * 8;
#pragma unroll
  for (int ti = 0; ti < 2; ++ti)
#pragma unroll
    for (int j = 0; j < 4; ++j)
#pragma unroll
      for (int v = 0; v < 8; ++v) {
        int row = m0 + wm * 32 + ti * 16 + mofs + v;
        int col = n0 + wn * 64 + j * 16 + nloc;
        float val = acc[ti][j][v] + bias[col];
        int b = row >> 11, s = row & (SEQ - 1);
        int h = col >> 6,  d = col & (HD - 1);
        size_t bh = (size_t)(b * HEADS + h);
        if (z == 0) {
          val *= 0.125f;   // HEAD_DIM^-0.5
          qbuf[(bh * SEQ + s) * HD + d] = f2bf(val);
        } else if (z == 1) {
          kbuf[(bh * SEQ + s) * HD + d] = f2bf(val);
        } else {
          vtbuf[(bh * HD + d) * SEQ + s] = f2bf(val);  // V transposed
        }
      }
}

// ---------------------------------------------------------------------------
// Kernel 2: flash attention. Block = 128 thr = 4 waves; each wave owns a
// 16-query tile; 32-wide KV tiles staged in LDS by the Tensor Data Mover.
// ---------------------------------------------------------------------------
__global__ __launch_bounds__(128) void attn_kernel(
    const unsigned short* __restrict__ qbuf,
    const unsigned short* __restrict__ kbuf,
    const unsigned short* __restrict__ vtbuf,
    const float* __restrict__ mask,
    unsigned short* __restrict__ ctxbuf) {
  __shared__ unsigned short kt[32 * 64];      // K tile   [kv][d]
  __shared__ unsigned short vt[64 * 32];      // V^T tile [d][kv]
  __shared__ unsigned short pt[4][16 * 32];   // per-wave P tile

  const int bh   = blockIdx.x;
  const int b    = bh >> 4, h = bh & 15;
  const int wave = threadIdx.x >> 5;
  const int lane = threadIdx.x & 31;
  const int q0   = blockIdx.y * 64 + wave * 16;

  const unsigned short* Qg = qbuf + ((size_t)bh * SEQ + q0) * HD;
  v16bf qa0 = load_frag<64>(Qg, 0);
  v16bf qa1 = load_frag<64>(Qg + 32, 0);

  float rowmax[8], rowsum[8];
  v8f zero = {};
  v8f cacc[4];
#pragma unroll
  for (int v = 0; v < 8; ++v) { rowmax[v] = -1e30f; rowsum[v] = 0.0f; }
#pragma unroll
  for (int g = 0; g < 4; ++g) cacc[g] = zero;

  for (int kv0 = 0; kv0 < SEQ; kv0 += 32) {
    __syncthreads();
#if USE_TDM
    if (wave == 0) {
      // K tensor per head: [2048, 64] row-major; tile = 32 rows x 64.
      tdm_load_2d_bf16(lds_offset_of(kt),
                       kbuf + ((size_t)bh * SEQ + kv0) * HD,
                       HD, SEQ, HD, HD, 32);
      // V^T tensor per head: [64, 2048]; tile = 64 rows x 32 cols at kv0.
      tdm_load_2d_bf16(lds_offset_of(vt),
                       vtbuf + (size_t)bh * HD * SEQ + kv0,
                       SEQ, HD, SEQ, 32, HD);
      __builtin_amdgcn_s_wait_tensorcnt(0);
    }
#else
#pragma unroll
    for (int it = 0; it < 2; ++it) {
      int idx = threadIdx.x + it * 128;
      int r = idx >> 3, c = (idx & 7) * 8;
      *(uint4*)(kt + r * 64 + c) =
          *(const uint4*)(kbuf + ((size_t)bh * SEQ + kv0 + r) * HD + c);
      int d = idx >> 2, c2 = (idx & 3) * 8;
      *(uint4*)(vt + d * 32 + c2) =
          *(const uint4*)(vtbuf + ((size_t)bh * HD + d) * SEQ + kv0 + c2);
    }
#endif
    __syncthreads();

    // S = Q * K^T  (two 16x16 tiles covering 32 kv columns)
    v8f s0 = zero, s1 = zero;
    v16bf kb;
    kb = load_frag<64>(kt, 0);        s0 = wmma_bf16(qa0, kb, s0);
    kb = load_frag<64>(kt + 32, 0);   s0 = wmma_bf16(qa1, kb, s0);
    kb = load_frag<64>(kt, 16);       s1 = wmma_bf16(qa0, kb, s1);
    kb = load_frag<64>(kt + 32, 16);  s1 = wmma_bf16(qa1, kb, s1);

    const float mv0 = mask[b * SEQ + kv0 + (lane & 15)];
    const float mv1 = mask[b * SEQ + kv0 + 16 + (lane & 15)];
    const int mrow  = ((lane >> 4) & 1) * 8;

#pragma unroll
    for (int v = 0; v < 8; ++v) {
      float e0 = s0[v] + mv0;
      float e1 = s1[v] + mv1;
      float mx = fmaxf(e0, e1);
#pragma unroll
      for (int x = 8; x >= 1; x >>= 1) mx = fmaxf(mx, __shfl_xor(mx, x, 32));
      float nm = fmaxf(rowmax[v], mx);
      float sc = __expf(rowmax[v] - nm);
      rowmax[v] = nm;
      rowsum[v] *= sc;
#pragma unroll
      for (int g = 0; g < 4; ++g) cacc[g][v] *= sc;
      float p0 = __expf(e0 - nm);
      float p1 = __expf(e1 - nm);
      float ls = p0 + p1;
#pragma unroll
      for (int x = 8; x >= 1; x >>= 1) ls += __shfl_xor(ls, x, 32);
      rowsum[v] += ls;
      pt[wave][(mrow + v) * 32 + (lane & 15)]      = f2bf(p0);
      pt[wave][(mrow + v) * 32 + 16 + (lane & 15)] = f2bf(p1);
    }
    asm volatile("s_wait_dscnt 0" ::: "memory");   // wave-local P round-trip

    v16bf pf = load_frag<32>(pt[wave], 0);
#pragma unroll
    for (int g = 0; g < 4; ++g) {
      v16bf vb = load_frag<32>(vt, g * 16);
      cacc[g] = wmma_bf16(pf, vb, cacc[g]);
    }
  }

#pragma unroll
  for (int g = 0; g < 4; ++g)
#pragma unroll
    for (int v = 0; v < 8; ++v) {
      int m = ((lane >> 4) & 1) * 8 + v;
      int n = lane & 15;
      float val = cacc[g][v] / rowsum[v];
      int s   = q0 + m;
      int col = h * HD + g * 16 + n;
      ctxbuf[((size_t)b * SEQ + s) * HID + col] = f2bf(val);
    }
}

// ---------------------------------------------------------------------------
// Kernel 3: output projection. ctx(bf16)[8192,1024] x Wo(fp32->bf16) + bo
// -> fp32 d_out.  A tile staged via TDM (already bf16, no conversion).
// ---------------------------------------------------------------------------
__global__ __launch_bounds__(256) void oproj_kernel(
    const unsigned short* __restrict__ ctx,
    const float* __restrict__ Wo, const float* __restrict__ bo,
    float* __restrict__ out) {
  __shared__ unsigned short sA[128 * 32];
  __shared__ unsigned short sB[128 * 32];
  const int m0 = blockIdx.x * 128;
  const int n0 = blockIdx.y * 128;
  const int tid  = threadIdx.x;
  const int wave = tid >> 5;
  const int wm   = wave & 3;
  const int wn   = wave >> 2;

  v8f zero = {};
  v8f acc[2][4];
#pragma unroll
  for (int i = 0; i < 2; ++i)
#pragma unroll
    for (int j = 0; j < 4; ++j) acc[i][j] = zero;

  for (int k0 = 0; k0 < HID; k0 += 32) {
    __syncthreads();
#if USE_TDM
    if (wave == 0) {
      // ctx tensor [8192, 1024]; tile = 128 rows x 32 cols at (m0, k0).
      tdm_load_2d_bf16(lds_offset_of(sA),
                       ctx + (size_t)m0 * HID + k0,
                       HID, NB * SEQ, HID, 32, 128);
    }
#else
    {
      int r = tid >> 1, c = (tid & 1) * 16;
      const unsigned short* src = ctx + (size_t)(m0 + r) * HID + k0 + c;
      *(uint4*)(sA + r * 32 + c)     = *(const uint4*)(src);
      *(uint4*)(sA + r * 32 + c + 8) = *(const uint4*)(src + 8);
    }
#endif
    {   // Wo tile transposed, fp32 -> bf16
      int k = tid >> 3, nb = (tid & 7) * 16;
      const float* src = Wo + (size_t)(k0 + k) * HID + n0 + nb;
      if (k0 + 32 < HID) __builtin_prefetch(src + 32 * HID, 0, 1);
#pragma unroll
      for (int i = 0; i < 16; ++i) sB[(nb + i) * 32 + k] = f2bf(src[i]);
    }
#if USE_TDM
    if (wave == 0) __builtin_amdgcn_s_wait_tensorcnt(0);
#endif
    __syncthreads();
    v16bf a0 = load_frag<32>(sA, wm * 32);
    v16bf a1 = load_frag<32>(sA, wm * 32 + 16);
#pragma unroll
    for (int j = 0; j < 4; ++j) {
      v16bf bf = load_frag<32>(sB, wn * 64 + j * 16);
      acc[0][j] = wmma_bf16(a0, bf, acc[0][j]);
      acc[1][j] = wmma_bf16(a1, bf, acc[1][j]);
    }
  }

  const int lane = tid & 31;
  const int nloc = lane & 15;
  const int mofs = ((lane >> 4) & 1) * 8;
#pragma unroll
  for (int ti = 0; ti < 2; ++ti)
#pragma unroll
    for (int j = 0; j < 4; ++j)
#pragma unroll
      for (int v = 0; v < 8; ++v) {
        int row = m0 + wm * 32 + ti * 16 + mofs + v;
        int col = n0 + wn * 64 + j * 16 + nloc;
        out[(size_t)row * HID + col] = acc[ti][j][v] + bo[col];
      }
}

extern "C" void kernel_launch(void* const* d_in, const int* in_sizes, int n_in,
                              void* d_out, int out_size, void* d_ws, size_t ws_size,
                              hipStream_t stream) {
  const float* batch = (const float*)d_in[0];
  const float* mask  = (const float*)d_in[1];
  const float* Wq = (const float*)d_in[2];
  const float* bq = (const float*)d_in[3];
  const float* Wk = (const float*)d_in[4];
  const float* bk = (const float*)d_in[5];
  const float* Wv = (const float*)d_in[6];
  const float* bv = (const float*)d_in[7];
  const float* Wo = (const float*)d_in[8];
  const float* bo = (const float*)d_in[9];
  float* out = (float*)d_out;

  const size_t N = (size_t)NB * SEQ * HID;   // 8,388,608 elems per buffer
  unsigned short* qbuf   = (unsigned short*)d_ws;
  unsigned short* kbuf   = qbuf + N;
  unsigned short* vtbuf  = kbuf + N;
  unsigned short* ctxbuf = vtbuf + N;

  dim3 g1(NB * SEQ / 128, HID / 128, 3);
  qkv_kernel<<<g1, 256, 0, stream>>>(batch, Wq, bq, Wk, bk, Wv, bv,
                                     qbuf, kbuf, vtbuf);

  dim3 g2(NB * HEADS, SEQ / 64);
  attn_kernel<<<g2, 128, 0, stream>>>(qbuf, kbuf, vtbuf, mask, ctxbuf);

  dim3 g3(NB * SEQ / 128, HID / 128);
  oproj_kernel<<<g3, 256, 0, stream>>>(ctxbuf, Wo, bo, out);
}
